// AttentionLayer_3435973837471
// MI455X (gfx1250) — compile-verified, matrix operand order checked by reading, not run on hardware
//
#include <hip/hip_runtime.h>
#include <hip/hip_bf16.h>

// Problem constants (from reference)
#define BB 64
#define SS 4096
#define HH 512
#define HD 256            // hidden dim of scorer MLP
#define ROWS (BB*SS)      // 262144 rows of X

typedef __attribute__((ext_vector_type(16))) __bf16 v16bf;
typedef __attribute__((ext_vector_type(8)))  float  v8f;
typedef __attribute__((ext_vector_type(4)))  unsigned int u32x4;
typedef __attribute__((ext_vector_type(8)))  int          i32x8;
typedef __attribute__((ext_vector_type(4)))  int          i32x4;

#if defined(__has_builtin)
#if __has_builtin(__builtin_amdgcn_tensor_load_to_lds) && __has_builtin(__builtin_amdgcn_s_wait_tensorcnt)
#define HAVE_TDM 1
#endif
#endif
#ifndef HAVE_TDM
#define HAVE_TDM 0
#endif

union FragU { uint4 u[2]; v16bf v; };

__device__ __forceinline__ unsigned short f2bf(float f) {
    unsigned u = __float_as_uint(f);
    unsigned r = u + 0x7FFFu + ((u >> 16) & 1u);   // round-to-nearest-even
    return (unsigned short)(r >> 16);
}

#if HAVE_TDM
// Issue a TDM DMA of one contiguous 16KB W1t slice (8192 bf16) into LDS.
// D# per CDNA5 ISA ch.8: group0 = {count/flags, lds_addr, global_addr lo/hi+type},
// group1 = {data_size, tensor dims/strides, tile dims}. 1-D tile, data_size=2B.
// This toolchain's builtin takes 6 args (g0, g1, g2, g3, g4, cpol).
__device__ __forceinline__ void tdm_load_b_slice(const unsigned short* gsrc,
                                                 unsigned lds_byte_addr) {
    unsigned long long ga = (unsigned long long)(uintptr_t)gsrc;
    u32x4 g0;
    g0[0] = 1u;                                          // count=1, user D#
    g0[1] = lds_byte_addr;                               // lds_addr[31:0]
    g0[2] = (unsigned)(ga & 0xFFFFFFFFu);                // global_addr[31:0]
    g0[3] = (unsigned)((ga >> 32) & 0x01FFFFFFu)         // global_addr[56:32]
          | (2u << 30);                                  // type = 2 ("image")
    i32x8 g1;
    g1[0] = (int)(1u << 16);          // data_size = 1 -> 2 bytes; no multicast
    g1[1] = 0;                        // tensor_dim0[15:0] (131072 -> low16 = 0)
    g1[2] = (int)((1u << 16) | 2u);   // tensor_dim0[31:16]=2, tensor_dim1=1
    g1[3] = (int)(8192u << 16);       // tile_dim0 = 8192 elements
    g1[4] = 0;                        // tile_dim1 = 0 (unused), tile_dim2 = 0
    g1[5] = (int)131072u;             // tensor_dim0_stride[31:0]
    g1[6] = 0;                        // stride hi, dim1_stride lo
    g1[7] = 0;
    i32x4 z4 = {0, 0, 0, 0};
    i32x8 z8 = {0, 0, 0, 0, 0, 0, 0, 0};
    __builtin_amdgcn_tensor_load_to_lds(g0, g1, z4, z4, z8, 0);
}
#endif

// ---------------------------------------------------------------------------
// Kernel 0: W1 (HH x HD f32, row-major) -> W1t (HD x HH bf16, row-major in k)
// ---------------------------------------------------------------------------
__global__ void attn_prep_w1t(const float* __restrict__ W1,
                              unsigned short* __restrict__ w1t) {
    int idx = blockIdx.x * blockDim.x + threadIdx.x;   // over HD*HH = 131072
    int n = idx >> 9;          // 0..255
    int k = idx & 511;         // 0..511
    w1t[idx] = f2bf(W1[(size_t)k * HD + n]);
}

// ---------------------------------------------------------------------------
// Kernel 1: scores[r] = b2 + sum_n W2[n]*tanh( X[r,:] . W1[:,n] + b1[n] )
// 8 waves/block, each wave owns one 16-row M-tile; W1t N-slices double-
// buffered in LDS via the Tensor Data Mover, overlapped with WMMA compute.
// ---------------------------------------------------------------------------
#define WAVES 8
__global__ __launch_bounds__(WAVES * 32)
void attn_scores_wmma(const float* __restrict__ X,
                      const unsigned short* __restrict__ w1t,
                      const float* __restrict__ b1,
                      const float* __restrict__ W2,
                      const float* __restrict__ b2,
                      float* __restrict__ scores) {
    __shared__ unsigned short ldsA[WAVES][16 * 512];   // per-wave A tile, bf16
    __shared__ unsigned short ldsB[2][16 * 512];       // double-buffered B tile

    const int tid  = threadIdx.x;
    const int w    = tid >> 5;
    const int lane = tid & 31;
    const int tile = blockIdx.x * WAVES + w;           // M-tile id (16 rows)

    // Wave-uniform "am I wave 0" predicate -> scalar branch around TDM issue
    const bool wave0 = (__builtin_amdgcn_readfirstlane(tid) < 32);

#if HAVE_TDM
    if (wave0) {
        // Kick off DMA of B slice 0 while everyone converts the A tiles.
        tdm_load_b_slice(w1t, (unsigned)(uintptr_t)&ldsB[0][0]);
    }
#endif

    // ---- Stage A: 16 rows x 512 f32 -> bf16 in LDS (linear copy, fp32->bf16)
    const float* src = X + (size_t)tile * (16 * 512);
    #pragma unroll 4
    for (int it = 0; it < 64; ++it) {
        int idx = it * 128 + lane * 4;
        float4 f = *(const float4*)(src + idx);
        uint2 pk;
        pk.x = (unsigned)f2bf(f.x) | ((unsigned)f2bf(f.y) << 16);
        pk.y = (unsigned)f2bf(f.z) | ((unsigned)f2bf(f.w) << 16);
        *(uint2*)&ldsA[w][idx] = pk;
    }

#if !HAVE_TDM
    { // Fallback: synchronous staged copy of B slice 0
        const unsigned short* bsrc = w1t;
        #pragma unroll
        for (int it = 0; it < 4; ++it) {
            int idx = it * 2048 + tid * 8;
            *(uint4*)&ldsB[0][idx] = *(const uint4*)(bsrc + idx);
        }
    }
#else
    if (wave0) __builtin_amdgcn_s_wait_tensorcnt(0);
#endif
    __syncthreads();

    const int mr   = lane & 15;     // row (A) / col (B) within 16x16 tile
    const int half = lane >> 4;

    float sacc[8] = {0.f, 0.f, 0.f, 0.f, 0.f, 0.f, 0.f, 0.f};

    for (int nt = 0; nt < 16; ++nt) {
        const int cur = nt & 1;

#if HAVE_TDM
        if (wave0 && nt + 1 < 16) {
            // Prefetch next B slice into the other buffer while we compute.
            tdm_load_b_slice(w1t + (nt + 1) * (16 * 512),
                             (unsigned)(uintptr_t)&ldsB[cur ^ 1][0]);
        }
#else
        if (nt + 1 < 16) {
            const unsigned short* bsrc = w1t + (nt + 1) * (16 * 512);
            #pragma unroll
            for (int it = 0; it < 4; ++it) {
                int idx = it * 2048 + tid * 8;
                *(uint4*)&ldsB[cur ^ 1][idx] = *(const uint4*)(bsrc + idx);
            }
        }
#endif

        v8f acc = {};
        #pragma unroll 4
        for (int kt = 0; kt < 16; ++kt) {
            // A fragment: lane holds row mr, K = kt*32 + half*8 + {0..7, 16..23}
            FragU a, b;
            int fbase = mr * 512 + kt * 32 + half * 8;
            a.u[0] = *(const uint4*)&ldsA[w][fbase];
            a.u[1] = *(const uint4*)&ldsA[w][fbase + 16];
            // B fragment: lane holds column mr, same K pattern (W1t row-major in k)
            b.u[0] = *(const uint4*)&ldsB[cur][fbase];
            b.u[1] = *(const uint4*)&ldsB[cur][fbase + 16];
            acc = __builtin_amdgcn_wmma_f32_16x16x32_bf16(
                false, a.v, false, b.v, (short)0, acc, false, false);
        }

        // Epilogue for this N-tile: h = tanh(acc + b1), score += h * W2
        // C layout: VGPR r, lane l -> M = r + 8*half, N = l&15
        int n = nt * 16 + mr;
        float b1v = b1[n];
        float w2v = W2[n];
        #pragma unroll
        for (int r = 0; r < 8; ++r)
            sacc[r] += tanhf(acc[r] + b1v) * w2v;

#if HAVE_TDM
        if (wave0 && nt + 1 < 16) __builtin_amdgcn_s_wait_tensorcnt(0);
#endif
        __syncthreads();   // publish next buffer / allow overwrite of cur
    }

    // Reduce across the 16 lanes (same M) within each half of the wave
    #pragma unroll
    for (int r = 0; r < 8; ++r) {
        float v = sacc[r];
        v += __shfl_xor(v, 1, 32);
        v += __shfl_xor(v, 2, 32);
        v += __shfl_xor(v, 4, 32);
        v += __shfl_xor(v, 8, 32);
        sacc[r] = v;
    }
    if (mr == 0) {
        float bias = b2[0];
        #pragma unroll
        for (int r = 0; r < 8; ++r)
            scores[(size_t)tile * 16 + r + 8 * half] = sacc[r] + bias;
    }
}

// ---------------------------------------------------------------------------
// Kernel 2: per-batch softmax stats (max, sum of exp)
// ---------------------------------------------------------------------------
__global__ void attn_softmax_stats(const float* __restrict__ scores,
                                   float* __restrict__ stats) {
    __shared__ float red[256];
    int b = blockIdx.x, tid = threadIdx.x;
    const float* s = scores + (size_t)b * SS;

    float m = -INFINITY;
    for (int i = tid; i < SS; i += 256) m = fmaxf(m, s[i]);
    red[tid] = m; __syncthreads();
    for (int o = 128; o > 0; o >>= 1) {
        if (tid < o) red[tid] = fmaxf(red[tid], red[tid + o]);
        __syncthreads();
    }
    m = red[0]; __syncthreads();

    float sum = 0.f;
    for (int i = tid; i < SS; i += 256) sum += expf(s[i] - m);
    red[tid] = sum; __syncthreads();
    for (int o = 128; o > 0; o >>= 1) {
        if (tid < o) red[tid] += red[tid + o];
        __syncthreads();
    }
    if (tid == 0) { stats[b * 2] = m; stats[b * 2 + 1] = red[0]; }
}

// ---------------------------------------------------------------------------
// Kernel 3: attn = exp(s-m)/l (written out), context = sum_s attn * X[b,s,:]
// One block per batch, 128 threads, each owns 4 consecutive h (float4).
// ---------------------------------------------------------------------------
__global__ __launch_bounds__(128)
void attn_context(const float* __restrict__ X,
                  const float* __restrict__ scores,
                  const float* __restrict__ stats,
                  float* __restrict__ attn,
                  float* __restrict__ ctx) {
    __shared__ float p[128];
    int b = blockIdx.x, tid = threadIdx.x;
    float m    = stats[b * 2];
    float invl = 1.0f / stats[b * 2 + 1];
    const float4* X4  = (const float4*)X;
    const float* srow = scores + (size_t)b * SS;
    float*       arow = attn   + (size_t)b * SS;

    float4 acc = {0.f, 0.f, 0.f, 0.f};
    for (int sc = 0; sc < SS; sc += 128) {
        float pv = expf(srow[sc + tid] - m) * invl;
        p[tid] = pv;
        arow[sc + tid] = pv;
        __syncthreads();

        size_t base = ((size_t)(b * SS + sc)) * 128 + tid;
        if (sc + 128 < SS)
            __builtin_prefetch((const void*)(X4 + base + 128 * 128), 0, 1);
        for (int i = 0; i < 128; ++i) {
            float4 x = X4[base + (size_t)i * 128];
            float  wv = p[i];
            acc.x += wv * x.x; acc.y += wv * x.y;
            acc.z += wv * x.z; acc.w += wv * x.w;
        }
        __syncthreads();
    }
    ((float4*)ctx)[(size_t)b * 128 + tid] = acc;
}

// ---------------------------------------------------------------------------
extern "C" void kernel_launch(void* const* d_in, const int* in_sizes, int n_in,
                              void* d_out, int out_size, void* d_ws, size_t ws_size,
                              hipStream_t stream) {
    const float* X  = (const float*)d_in[0];   // (B,S,H)
    const float* W1 = (const float*)d_in[1];   // (H,HD)
    const float* b1 = (const float*)d_in[2];   // (HD)
    const float* W2 = (const float*)d_in[3];   // (HD,1)
    const float* b2 = (const float*)d_in[4];   // (1)

    float* out_ctx  = (float*)d_out;                   // (B,H)   = 32768 floats
    float* out_attn = out_ctx + (size_t)BB * HH;       // (B,S,1) = 262144 floats

    // Workspace layout
    char* ws = (char*)d_ws;
    float* scores = (float*)ws;                              // ROWS floats (1 MiB)
    float* stats  = scores + ROWS;                           // 2*BB floats
    unsigned short* w1t = (unsigned short*)(ws + (size_t)ROWS * 4 + 2 * BB * 4);
    // w1t: HD*HH bf16 = 256 KB

    attn_prep_w1t<<<(HD * HH) / 256, 256, 0, stream>>>(W1, w1t);
    attn_scores_wmma<<<ROWS / (16 * WAVES), WAVES * 32, 0, stream>>>(
        X, w1t, b1, W2, b2, scores);
    attn_softmax_stats<<<BB, 256, 0, stream>>>(scores, stats);
    attn_context<<<BB, 128, 0, stream>>>(X, scores, stats, out_attn, out_ctx);
}
